// SwinTransformerBlock_38826504356437
// MI455X (gfx1250) — matrix operand order repaired and checked by reading, not compiled, not run on hardware
//
#include <hip/hip_runtime.h>
#include <math.h>

// ---------------------------------------------------------------------------
// Swin block on gfx1250: all matmuls via v_wmma_f32_16x16x32_f16 (wave32),
// GEMM staging via double-buffered GLOBAL_LOAD_ASYNC_TO_LDS_B128 (ASYNCcnt).
// ---------------------------------------------------------------------------

typedef __attribute__((ext_vector_type(16))) _Float16    v16h;
typedef __attribute__((ext_vector_type(8)))  float       v8f;
typedef __attribute__((ext_vector_type(8)))  unsigned int v8u;
typedef __attribute__((ext_vector_type(4)))  int         v4i;

union FragU { v8u u; v16h h; };

constexpr int DIM    = 512;
constexpr int NHEAD  = 16;
constexpr int HD     = 32;     // head dim
constexpr int QKVN   = 1536;   // 3*DIM
constexpr int MLPN   = 2048;
constexpr int TOK    = 32768;  // 8 * 64 * 64 tokens
constexpr int NWIN   = 512;    // 8 images * 64 windows
constexpr float QK_SCALE = 0.17677669529663687f; // 32^-0.5

#if defined(__gfx1250__) && __has_builtin(__builtin_amdgcn_global_load_async_to_lds_b128)
#define USE_ASYNC_LDS 1
#else
#define USE_ASYNC_LDS 0
#endif

// builtin signature (from compiler diagnostic): (v4i AS(1)*, v4i AS(3)*, imm, imm)
typedef __attribute__((address_space(1))) v4i* g4_t;
typedef __attribute__((address_space(3))) v4i* l4_t;

// dest windowed-token -> source image-token (cyclic shift -4,-4 + window partition)
// Same mapping is the scatter target for window-reverse + shift +4,+4.
__device__ __forceinline__ int src_index(int t) {
  int w  = t >> 6, n = t & 63;
  int b  = w >> 6, wi = w & 63;
  int sh = ((wi >> 3) << 3) + (n >> 3);
  int sw = ((wi & 7) << 3) + (n & 7);
  int h  = (sh + 4) & 63;
  int c  = (sw + 4) & 63;
  return (b << 12) + (h << 6) + c;
}

// shift-mask region id of token n inside window at (wrow, wcol) of the shifted canvas
__device__ __forceinline__ int tok_region(int wrow, int wcol, int n) {
  int sh = wrow + (n >> 3), sw = wcol + (n & 7);
  int rr = (sh < 56) ? 0 : ((sh < 60) ? 1 : 2);
  int cc = (sw < 56) ? 0 : ((sw < 60) ? 1 : 2);
  return rr * 3 + cc;
}

// ---------------------------------------------------------------------------
// f32 [K,N] weight -> f16 transposed [N,K] (LDS-tiled, coalesced both sides).
__global__ __launch_bounds__(256)
void cvt_transpose(const float* __restrict__ in, _Float16* __restrict__ out,
                   int K, int N) {
  __shared__ float t[32][33];
  int tx = threadIdx.x & 31, ty = threadIdx.x >> 5;  // 32x8
  int k0 = blockIdx.x * 32, n0 = blockIdx.y * 32;
#pragma unroll
  for (int s = 0; s < 4; ++s)
    t[ty + s * 8][tx] = in[(size_t)(k0 + ty + s * 8) * N + n0 + tx];
  __syncthreads();
#pragma unroll
  for (int s = 0; s < 4; ++s)
    out[(size_t)(n0 + ty + s * 8) * K + k0 + tx] = (_Float16)t[tx][ty + s * 8];
}

// biasNH[h][n][m] = rel_bias_table[rel_idx(n,m)][h]   (16*64*64 f32 = 256KB)
__global__ void build_bias(const float* __restrict__ tbl,
                           float* __restrict__ biasNH) {
  int e = blockIdx.x * 256 + threadIdx.x;   // 65536 total
  int h = e >> 12, rem = e & 4095, n = rem >> 6, m = rem & 63;
  int idx = ((n >> 3) - (m >> 3) + 7) * 15 + ((n & 7) - (m & 7) + 7);
  biasNH[e] = tbl[idx * NHEAD + h];
}

// ---------------------------------------------------------------------------
// LayerNorm over DIM=512, one token per 256-thread block, f16 output.
// GATHER: output index = windowed token, input row = shifted source token.
template <bool GATHER>
__global__ __launch_bounds__(256)
void ln_kernel(const float* __restrict__ x, const float* __restrict__ g,
               const float* __restrict__ bta, _Float16* __restrict__ out) {
  __shared__ float r1[256], r2[256];
  int t   = blockIdx.x;
  int row = GATHER ? src_index(t) : t;
  int tid = threadIdx.x;
  const float* xp = x + (size_t)row * DIM;
  int i0 = tid * 2;
  float a = xp[i0], b = xp[i0 + 1];
  r1[tid] = a + b;
  r2[tid] = a * a + b * b;
  __syncthreads();
  for (int s = 128; s > 0; s >>= 1) {
    if (tid < s) { r1[tid] += r1[tid + s]; r2[tid] += r2[tid + s]; }
    __syncthreads();
  }
  float mean = r1[0] * (1.0f / DIM);
  float var  = r2[0] * (1.0f / DIM) - mean * mean;
  float rs   = rsqrtf(var + 1e-5f);
  _Float16* op = out + (size_t)t * DIM;
  op[i0]     = (_Float16)((a - mean) * rs * g[i0]     + bta[i0]);
  op[i0 + 1] = (_Float16)((b - mean) * rs * g[i0 + 1] + bta[i0 + 1]);
}

// ---------------------------------------------------------------------------
// Tiled WMMA GEMM:  C[M,N] = A[M,K](f16) * B^T[N,K](f16) + bias, fused epilogue.
// 128x128 tile / workgroup, 8 waves, each wave owns 64x32 (4x2 16x16 tiles).
// Double-buffered async global->LDS staging tracked with ASYNCcnt.
// MODE 0: QKV   -> f16 out [row*1536+col], scale q (col<512) by 32^-0.5
// MODE 1: PROJ  -> f32 out scattered to d_out[src_index(row)] + residual x
// MODE 2: FC1   -> f16 out [row*2048+col], exact GELU
// MODE 3: FC2   -> f32 accumulate into d_out[row*512+col]
constexpr int BM = 128, BN = 128, BK = 32;
constexpr int LDP = 40;  // LDS row pitch in halves (80B: b128-aligned, odd bank stride)

template <int MODE>
__global__ __launch_bounds__(256)
void gemm_wmma(const _Float16* __restrict__ A, const _Float16* __restrict__ Bt,
               const float* __restrict__ bias, int K,
               _Float16* __restrict__ outh, float* __restrict__ outf,
               const float* __restrict__ resid) {
  __shared__ _Float16 As[2][BM][LDP];
  __shared__ _Float16 Bs[2][BN][LDP];  // Bs[n][k] (Bt already transposed)
  int tid   = threadIdx.x;
  int mBase = blockIdx.y * BM;
  int nBase = blockIdx.x * BN;
  int wave  = tid >> 5, lane = tid & 31;
  int wm = wave >> 2, wn = wave & 3;        // 2x4 wave grid
  int lm = lane & 15, hf = lane >> 4;

  // each thread copies one b128 (8 halves) for rows r0 and r0+64 of A and B
  int r0 = tid >> 2, kc0 = (tid & 3) * 8;
  const _Float16* Ar0 = A  + (size_t)(mBase + r0)      * K + kc0;
  const _Float16* Ar1 = A  + (size_t)(mBase + r0 + 64) * K + kc0;
  const _Float16* Br0 = Bt + (size_t)(nBase + r0)      * K + kc0;
  const _Float16* Br1 = Bt + (size_t)(nBase + r0 + 64) * K + kc0;

  auto issue_stage = [&](int kt, int buf) {
#if USE_ASYNC_LDS
    __builtin_amdgcn_global_load_async_to_lds_b128(
        (g4_t)(void*)(Ar0 + kt), (l4_t)&As[buf][r0][kc0], 0, 0);
    __builtin_amdgcn_global_load_async_to_lds_b128(
        (g4_t)(void*)(Ar1 + kt), (l4_t)&As[buf][r0 + 64][kc0], 0, 0);
    __builtin_amdgcn_global_load_async_to_lds_b128(
        (g4_t)(void*)(Br0 + kt), (l4_t)&Bs[buf][r0][kc0], 0, 0);
    __builtin_amdgcn_global_load_async_to_lds_b128(
        (g4_t)(void*)(Br1 + kt), (l4_t)&Bs[buf][r0 + 64][kc0], 0, 0);
#else
    *(uint4*)&As[buf][r0][kc0]      = *(const uint4*)(Ar0 + kt);
    *(uint4*)&As[buf][r0 + 64][kc0] = *(const uint4*)(Ar1 + kt);
    *(uint4*)&Bs[buf][r0][kc0]      = *(const uint4*)(Br0 + kt);
    *(uint4*)&Bs[buf][r0 + 64][kc0] = *(const uint4*)(Br1 + kt);
#endif
  };

  v8f zero = {0.f, 0.f, 0.f, 0.f, 0.f, 0.f, 0.f, 0.f};
  v8f acc[4][2];
#pragma unroll
  for (int i = 0; i < 4; ++i)
#pragma unroll
    for (int j = 0; j < 2; ++j) acc[i][j] = zero;

  issue_stage(0, 0);
  int nstage = K / BK;
  for (int s = 0; s < nstage; ++s) {
    int  buf  = s & 1;
    bool more = (s + 1 < nstage);
    if (more) issue_stage((s + 1) * BK, buf ^ 1);  // prefetch into other buffer
#if USE_ASYNC_LDS
    // async loads complete in order: <=4 outstanding => this stage's 4 are done
    if (more) asm volatile("s_wait_asynccnt 0x4" ::: "memory");
    else      asm volatile("s_wait_asynccnt 0x0" ::: "memory");
#endif
    __syncthreads();

    // B fragments (32x16): lane=column, K contiguous in transposed LDS
    FragU bf[2];
#pragma unroll
    for (int j = 0; j < 2; ++j) {
      const unsigned* bp =
          (const unsigned*)&Bs[buf][wn * 32 + j * 16 + lm][hf * 16];
#pragma unroll
      for (int u = 0; u < 8; ++u) bf[j].u[u] = bp[u];
    }
#pragma unroll
    for (int i = 0; i < 4; ++i) {
      // A fragment (16x32): lane=row, K contiguous {16*half .. 16*half+15}
      FragU af;
      const unsigned* ap = (const unsigned*)&As[buf][wm * 64 + i * 16 + lm][0];
#pragma unroll
      for (int u = 0; u < 4; ++u) {
        af.u[u]     = ap[u + 8 * hf];
        af.u[u + 4] = ap[4 + u + 8 * hf];
      }
#pragma unroll
      for (int j = 0; j < 2; ++j)
        acc[i][j] = __builtin_amdgcn_wmma_f32_16x16x32_f16(
            false, af.h, false, bf[j].h, (short)0, acc[i][j], false, false);
    }
    __syncthreads();  // all waves done reading buf before it is refilled
  }

  // Epilogue: D layout = vgpr r -> row r + 8*half, col = lane&15
#pragma unroll
  for (int i = 0; i < 4; ++i)
#pragma unroll
    for (int j = 0; j < 2; ++j) {
      int col  = nBase + wn * 32 + j * 16 + lm;
      float bv = bias[col];
#pragma unroll
      for (int r = 0; r < 8; ++r) {
        int row = mBase + wm * 64 + i * 16 + r + 8 * hf;
        float v = acc[i][j][r] + bv;
        if (MODE == 0) {
          if (col < DIM) v *= QK_SCALE;
          outh[(size_t)row * QKVN + col] = (_Float16)v;
        } else if (MODE == 1) {
          int s2 = src_index(row);
          size_t o = (size_t)s2 * DIM + col;
          outf[o] = resid[o] + v;
        } else if (MODE == 2) {
          float gl = 0.5f * v * (1.0f + erff(v * 0.70710678118f));
          outh[(size_t)row * MLPN + col] = (_Float16)gl;
        } else {
          size_t o = (size_t)row * DIM + col;
          outf[o] = outf[o] + v;
        }
      }
    }
}

// ---------------------------------------------------------------------------
// Attention: one block per (window, head), 4 waves. S = q k^T (+bias+mask),
// softmax, O = P v.  q scaled already at QKV store.
__global__ __launch_bounds__(128)
void attn_kernel(const _Float16* __restrict__ qkv,
                 const float* __restrict__ biasNH,
                 _Float16* __restrict__ outh) {
  __shared__ _Float16 qs[64][LDP];
  __shared__ _Float16 ks[64][LDP];
  __shared__ float    Sf[64][65];
  __shared__ _Float16 Pf[64][72];
  __shared__ _Float16 vT[32][72];   // v transposed: vT[d][token]
  __shared__ float    red[64][4];

  int tid = threadIdx.x;
  int blk = blockIdx.x;
  int w = blk >> 4, h = blk & 15;
  int wi = w & 63;
  int wrow = (wi >> 3) << 3, wcol = (wi & 7) << 3;
  const _Float16* base = qkv + (size_t)w * 64 * QKVN + h * HD;

  // stage q, k: 64x32 halves each (2 uint4 per thread)
#pragma unroll
  for (int l = 0; l < 2; ++l) {
    int u = tid * 2 + l;
    int r = u >> 2, c8 = (u & 3) * 8;
    *(uint4*)&qs[r][c8] = *(const uint4*)(base + (size_t)r * QKVN + c8);
    *(uint4*)&ks[r][c8] = *(const uint4*)(base + (size_t)r * QKVN + DIM + c8);
  }
  // stage v transposed
  {
    int r = tid >> 1, d0 = (tid & 1) * 16;
    const _Float16* vp = base + (size_t)r * QKVN + 2 * DIM + d0;
#pragma unroll
    for (int j = 0; j < 16; ++j) vT[d0 + j][r] = vp[j];
  }
  __syncthreads();

  int wv = tid >> 5, lane = tid & 31, lm = lane & 15, hf = lane >> 4;
  v8f zero = {0.f, 0.f, 0.f, 0.f, 0.f, 0.f, 0.f, 0.f};

  // ---- S = q @ k^T : wave wv handles rows [wv*16, wv*16+16)
  FragU aq;
  {
    const unsigned* ap = (const unsigned*)&qs[wv * 16 + lm][0];
#pragma unroll
    for (int u = 0; u < 4; ++u) {
      aq.u[u]     = ap[u + 8 * hf];
      aq.u[u + 4] = ap[4 + u + 8 * hf];
    }
  }
#pragma unroll
  for (int nt = 0; nt < 4; ++nt) {
    FragU bf;  // B[d][m'] = k[m'][d] -> column m' = lane, K=d contiguous
    const unsigned* bp = (const unsigned*)&ks[nt * 16 + lm][hf * 16];
#pragma unroll
    for (int u = 0; u < 8; ++u) bf.u[u] = bp[u];
    v8f s = __builtin_amdgcn_wmma_f32_16x16x32_f16(
        false, aq.h, false, bf.h, (short)0, zero, false, false);
    int col  = nt * 16 + lm;
    int creg = tok_region(wrow, wcol, col);
#pragma unroll
    for (int r = 0; r < 8; ++r) {
      int m = wv * 16 + r + 8 * hf;
      float v = s[r] + biasNH[(h << 12) + (m << 6) + col];
      if (tok_region(wrow, wcol, m) != creg) v -= 100.0f;
      Sf[m][col] = v;
    }
  }
  __syncthreads();

  // ---- softmax over rows (2 threads per row, 32 cols each)
  {
    int row = tid >> 1, part = tid & 1, c0 = part * 32;
    float mx = -1e30f;
#pragma unroll
    for (int j = 0; j < 32; ++j) mx = fmaxf(mx, Sf[row][c0 + j]);
    red[row][part] = mx;
    __syncthreads();
    float M = fmaxf(red[row][0], red[row][1]);
    float sum = 0.f;
#pragma unroll
    for (int j = 0; j < 32; ++j) {
      float e = __expf(Sf[row][c0 + j] - M);
      Sf[row][c0 + j] = e;
      sum += e;
    }
    red[row][2 + part] = sum;
    __syncthreads();
    float inv = 1.0f / (red[row][2] + red[row][3]);
#pragma unroll
    for (int j = 0; j < 32; ++j)
      Pf[row][c0 + j] = (_Float16)(Sf[row][c0 + j] * inv);
  }
  __syncthreads();

  // ---- O = P @ v : M=64, K=64 (2 k-steps), N=32 (2 tiles)
  v8f oacc[2] = {zero, zero};
#pragma unroll
  for (int kstep = 0; kstep < 2; ++kstep) {
    FragU af;
    const unsigned* ap = (const unsigned*)&Pf[wv * 16 + lm][kstep * 32];
#pragma unroll
    for (int u = 0; u < 4; ++u) {
      af.u[u]     = ap[u + 8 * hf];
      af.u[u + 4] = ap[4 + u + 8 * hf];
    }
#pragma unroll
    for (int nt = 0; nt < 2; ++nt) {
      FragU bf;
      const unsigned* bp =
          (const unsigned*)&vT[nt * 16 + lm][kstep * 32 + hf * 16];
#pragma unroll
      for (int u = 0; u < 8; ++u) bf.u[u] = bp[u];
      oacc[nt] = __builtin_amdgcn_wmma_f32_16x16x32_f16(
          false, af.h, false, bf.h, (short)0, oacc[nt], false, false);
    }
  }
  // store head-concat layout [token][h*32+d] for proj GEMM
#pragma unroll
  for (int nt = 0; nt < 2; ++nt) {
    int d = nt * 16 + lm;
#pragma unroll
    for (int r = 0; r < 8; ++r) {
      int m = wv * 16 + r + 8 * hf;
      outh[(size_t)(w * 64 + m) * DIM + h * HD + d] = (_Float16)oacc[nt][r];
    }
  }
}

// ---------------------------------------------------------------------------
extern "C" void kernel_launch(void* const* d_in, const int* in_sizes, int n_in,
                              void* d_out, int out_size, void* d_ws,
                              size_t ws_size, hipStream_t stream) {
  const float* x       = (const float*)d_in[0];
  const float* norm1_g = (const float*)d_in[2];
  const float* norm1_b = (const float*)d_in[3];
  const float* qkv_w   = (const float*)d_in[4];
  const float* qkv_b   = (const float*)d_in[5];
  const float* relt    = (const float*)d_in[6];
  const float* proj_w  = (const float*)d_in[7];
  const float* proj_b  = (const float*)d_in[8];
  const float* norm2_g = (const float*)d_in[9];
  const float* norm2_b = (const float*)d_in[10];
  const float* fc1_w   = (const float*)d_in[11];
  const float* fc1_b   = (const float*)d_in[12];
  const float* fc2_w   = (const float*)d_in[13];
  const float* fc2_b   = (const float*)d_in[14];
  float* out = (float*)d_out;
  char*  ws  = (char*)d_ws;
  const size_t MB = 1u << 20;

  // persistent region [0, 8MB): f16 transposed weights + bias table
  _Float16* qkvW16  = (_Float16*)ws;                 // [1536][512]
  _Float16* projW16 = qkvW16 + 512 * 1536;           // [512][512]
  _Float16* fc1W16  = projW16 + 512 * 512;           // [2048][512]
  _Float16* fc2W16  = fc1W16 + 512 * 2048;           // [512][2048]
  float*    biasNH  = (float*)(fc2W16 + 2048 * 512);
  // [8MB, 40MB): LN output (wA, reused as wH2)   [32MB]
  _Float16* bufA   = (_Float16*)(ws + 8 * MB);
  // [40MB, 168MB): QKV [96MB], later MLP hidden [128MB]
  _Float16* bufQKV = (_Float16*)(ws + 40 * MB);
  // [136MB, 168MB): attention output [32MB] (dead before FC1 writes here)
  _Float16* bufAtt = (_Float16*)(ws + 136 * MB);

  cvt_transpose<<<dim3(512 / 32, 1536 / 32), 256, 0, stream>>>(qkv_w, qkvW16, 512, 1536);
  cvt_transpose<<<dim3(512 / 32, 512 / 32),  256, 0, stream>>>(proj_w, projW16, 512, 512);
  cvt_transpose<<<dim3(512 / 32, 2048 / 32), 256, 0, stream>>>(fc1_w, fc1W16, 512, 2048);
  cvt_transpose<<<dim3(2048 / 32, 512 / 32), 256, 0, stream>>>(fc2_w, fc2W16, 2048, 512);
  build_bias<<<(NHEAD * 64 * 64) / 256, 256, 0, stream>>>(relt, biasNH);

  // LN1 + shift + window-partition gather -> f16
  ln_kernel<true><<<TOK, 256, 0, stream>>>(x, norm1_g, norm1_b, bufA);
  // QKV GEMM (q pre-scaled)
  gemm_wmma<0><<<dim3(QKVN / BN, TOK / BM), 256, 0, stream>>>(
      bufA, qkvW16, qkv_b, DIM, bufQKV, nullptr, nullptr);
  // windowed attention
  attn_kernel<<<NWIN * NHEAD, 128, 0, stream>>>(bufQKV, biasNH, bufAtt);
  // proj GEMM + window-reverse/unshift scatter + residual -> d_out (f32)
  gemm_wmma<1><<<dim3(DIM / BN, TOK / BM), 256, 0, stream>>>(
      bufAtt, projW16, proj_b, DIM, nullptr, out, x);
  // LN2 -> f16
  ln_kernel<false><<<TOK, 256, 0, stream>>>(out, norm2_g, norm2_b, bufA);
  // FC1 + GELU
  gemm_wmma<2><<<dim3(MLPN / BN, TOK / BM), 256, 0, stream>>>(
      bufA, fc1W16, fc1_b, DIM, bufQKV, nullptr, nullptr);
  // FC2 + residual accumulate into d_out
  gemm_wmma<3><<<dim3(DIM / BN, TOK / BM), 256, 0, stream>>>(
      bufQKV, fc2W16, fc2_b, MLPN, nullptr, out, nullptr);
}
